// DynsSolver_87832081203605
// MI455X (gfx1250) — matrix operand order repaired and checked by reading, not compiled, not run on hardware
//
#include <hip/hip_runtime.h>

typedef __attribute__((ext_vector_type(16))) _Float16 v16h;
typedef __attribute__((ext_vector_type(8)))  float    v8f;

#define N_STEPS 400
#define HID     256
#define TPB     512   // 16 waves of 32; wave w owns output rows [16w, 16w+16)

__device__ __forceinline__ float grid_at(int i) {
    return (i == N_STEPS) ? 4.0f : (float)i * 0.01f;
}

__global__ __launch_bounds__(TPB)
void DynsSolver_node_rk4(const float* __restrict__ x,
                         const float* __restrict__ W1, const float* __restrict__ b1,
                         const float* __restrict__ W2, const float* __restrict__ b2,
                         const float* __restrict__ W3, const float* __restrict__ b3,
                         const float* __restrict__ W4, const float* __restrict__ b4,
                         float* __restrict__ out)
{
    __shared__ float sW1[2 * HID];
    __shared__ float sb1[HID], sb2[HID], sb3[HID];
    __shared__ float sW4[HID * 2];
    __shared__ float sb4[2];
    // activations stored directly in f16: WMMA B-fragments load straight from
    // LDS as b128 with no per-read cvt on the critical path
    __shared__ __align__(16) _Float16 hA[HID];
    __shared__ __align__(16) _Float16 hB[HID];
    __shared__ float ys[(N_STEPS + 1) * 2];     // full trajectory (3.2 KB)
    __shared__ float yin[2], kst[4][2], ycur[2];
    __shared__ float red0[8], red1[8];

    const int tid    = threadIdx.x;
    const int wave   = tid >> 5;          // 0..15 -> row tile
    const int lane   = tid & 31;
    const int laneHi = (lane >= 16) ? 1 : 0;
    const int mr     = lane & 15;
    const int mglob  = wave * 16 + mr;    // output row this lane supplies to A

    // ---- one-time: small params to LDS ----
    for (int i = tid; i < 2 * HID;  i += TPB) sW1[i] = W1[i];
    for (int i = tid; i < HID;      i += TPB) { sb1[i] = b1[i]; sb2[i] = b2[i]; sb3[i] = b3[i]; }
    for (int i = tid; i < HID * 2;  i += TPB) sW4[i] = W4[i];
    if (tid < 2) sb4[tid] = b4[tid];
    if (tid == 0) { ycur[0] = x[0]; ycur[1] = x[1]; ys[0] = x[0]; ys[1] = x[1]; }

    // ---- one-time: W2/W3 -> f16 WMMA A-fragments, resident in VGPRs ----
    // A (16x32 f16) layout: lane = row M; element e -> K = e + 16*laneHi.
    // A[m][k] = W[k][m] (W row-major [fan_in][fan_out]) so D = Wt*h.
    v16h a2[8], a3[8];
    #pragma unroll
    for (int c = 0; c < 8; ++c) {
        #pragma unroll
        for (int e = 0; e < 16; ++e) {
            int k = c * 32 + laneHi * 16 + e;
            a2[c][e] = (_Float16)W2[k * HID + mglob];
            a3[c][e] = (_Float16)W3[k * HID + mglob];
        }
    }
    __syncthreads();

    // ---- persistent time loop: 400 RK4 (3/8 rule) steps ----
    for (int step = 0; step < N_STEPS; ++step) {
        const float dt = grid_at(step + 1) - grid_at(step);

        for (int s = 0; s < 4; ++s) {
            if (tid == 0) {
                #pragma unroll
                for (int j = 0; j < 2; ++j) {
                    float y = ycur[j];
                    if      (s == 0) yin[j] = y;
                    else if (s == 1) yin[j] = y + dt * kst[0][j] / 3.0f;
                    else if (s == 2) yin[j] = y + dt * (kst[1][j] - kst[0][j] / 3.0f);
                    else             yin[j] = y + dt * (kst[0][j] - kst[1][j] + kst[2][j]);
                }
            }
            __syncthreads();

            // layer 1: 2 -> 256 (K=2, VALU), store f16
            if (tid < HID) {
                float v = yin[0] * sW1[tid] + yin[1] * sW1[HID + tid] + sb1[tid];
                hA[tid] = (_Float16)(v > 0.0f ? v : 0.0f);
            }
            __syncthreads();

            // layer 2: 256 -> 256 via WMMA, weights resident in a2[]
            {
                v8f acc = {};
                #pragma unroll
                for (int c = 0; c < 8; ++c) {
                    // B (32x16 f16): lane = col N; element e -> K = e + 16*laneHi.
                    // h broadcast into every column; 16 consecutive halves = 2x ds_load_b128.
                    v16h bf;
                    #pragma unroll
                    for (int e = 0; e < 16; ++e)
                        bf[e] = hA[c * 32 + laneHi * 16 + e];
                    acc = __builtin_amdgcn_wmma_f32_16x16x32_f16(
                              false, a2[c], false, bf, (short)0, acc, false, false);
                }
                // D: lane = N, VGPR r holds M = r + 8*laneHi. Columns identical; N=0 writes.
                if (mr == 0) {
                    #pragma unroll
                    for (int r = 0; r < 8; ++r) {
                        int m = wave * 16 + laneHi * 8 + r;
                        float v = acc[r] + sb2[m];
                        hB[m] = (_Float16)(v > 0.0f ? v : 0.0f);
                    }
                }
            }
            __syncthreads();

            // layer 3: 256 -> 256 via WMMA, weights resident in a3[]
            {
                v8f acc = {};
                #pragma unroll
                for (int c = 0; c < 8; ++c) {
                    v16h bf;
                    #pragma unroll
                    for (int e = 0; e < 16; ++e)
                        bf[e] = hB[c * 32 + laneHi * 16 + e];
                    acc = __builtin_amdgcn_wmma_f32_16x16x32_f16(
                              false, a3[c], false, bf, (short)0, acc, false, false);
                }
                if (mr == 0) {
                    #pragma unroll
                    for (int r = 0; r < 8; ++r) {
                        int m = wave * 16 + laneHi * 8 + r;
                        float v = acc[r] + sb3[m];
                        hA[m] = (_Float16)(v > 0.0f ? v : 0.0f);
                    }
                }
            }
            __syncthreads();

            // layer 4: 256 -> 2 (wave shfl reduction, then 8 partials)
            {
                float p0 = 0.0f, p1 = 0.0f;
                if (tid < HID) {
                    float h = (float)hA[tid];
                    p0 = h * sW4[tid * 2 + 0];
                    p1 = h * sW4[tid * 2 + 1];
                }
                #pragma unroll
                for (int off = 16; off > 0; off >>= 1) {
                    p0 += __shfl_xor(p0, off, 32);
                    p1 += __shfl_xor(p1, off, 32);
                }
                if (lane == 0 && wave < 8) { red0[wave] = p0; red1[wave] = p1; }
            }
            __syncthreads();
            if (tid == 0) {
                float f0 = sb4[0], f1 = sb4[1];
                #pragma unroll
                for (int wv = 0; wv < 8; ++wv) { f0 += red0[wv]; f1 += red1[wv]; }
                kst[s][0] = f0; kst[s][1] = f1;
            }
            __syncthreads();
        }

        if (tid == 0) {
            #pragma unroll
            for (int j = 0; j < 2; ++j) {
                float ny = ycur[j] +
                    dt * (kst[0][j] + 3.0f * (kst[1][j] + kst[2][j]) + kst[3][j]) * 0.125f;
                ycur[j] = ny;
                ys[(step + 1) * 2 + j] = ny;
            }
        }
        __syncthreads();
    }

    // ---- interpolation tail: ts = linspace(0,4,50), output points 1..49 ----
    if (tid < 49) {
        int   i = tid + 1;
        float t = 4.0f * (float)i / 49.0f;
        // searchsorted(grid, t, 'left'): first idx with grid[idx] >= t
        int lo = 0, hi = N_STEPS + 1;
        while (lo < hi) {
            int mid = (lo + hi) >> 1;
            if (grid_at(mid) < t) lo = mid + 1; else hi = mid;
        }
        int idx = lo - 1;
        if (idx < 0) idx = 0;
        if (idx > N_STEPS - 1) idx = N_STEPS - 1;
        float g0 = grid_at(idx), g1 = grid_at(idx + 1);
        float frac = (t - g0) / (g1 - g0);
        #pragma unroll
        for (int j = 0; j < 2; ++j) {
            float a = ys[idx * 2 + j], b = ys[(idx + 1) * 2 + j];
            out[tid * 2 + j] = a + frac * (b - a);
        }
    }
}

extern "C" void kernel_launch(void* const* d_in, const int* in_sizes, int n_in,
                              void* d_out, int out_size, void* d_ws, size_t ws_size,
                              hipStream_t stream) {
    // setup_inputs order: x, w(unused), W1, b1, W2, b2, W3, b3, W4, b4
    const float* x  = (const float*)d_in[0];
    const float* W1 = (const float*)d_in[2];
    const float* b1 = (const float*)d_in[3];
    const float* W2 = (const float*)d_in[4];
    const float* b2 = (const float*)d_in[5];
    const float* W3 = (const float*)d_in[6];
    const float* b3 = (const float*)d_in[7];
    const float* W4 = (const float*)d_in[8];
    const float* b4 = (const float*)d_in[9];
    float* out = (float*)d_out;

    DynsSolver_node_rk4<<<dim3(1), dim3(TPB), 0, stream>>>(
        x, W1, b1, W2, b2, W3, b3, W4, b4, out);
}
